// Qwen3GatedDeltaNet_22187801051668
// MI455X (gfx1250) — matrix operand order repaired and checked by reading, not compiled, not run on hardware
//
#include <hip/hip_runtime.h>
#include <hip/hip_bf16.h>

typedef unsigned short u16;
typedef __attribute__((ext_vector_type(16))) __bf16 v16bf;
typedef __attribute__((ext_vector_type(8)))  float  v8f;

#define BSZ   4
#define SEQ   2048
#define HIDN  2048
#define NHEAD 32
#define DKH   64
#define DVH   64
#define NQ    8256   /* 8192 qkvz cols + 64 ba cols */
#define MROWS 8192   /* B*S */
#define EPSV  1e-6f
#define CHUNK 32     /* timesteps staged per async chunk in the recurrence */

__device__ __forceinline__ u16 f32_to_bf16(float f) {
    unsigned int u = __float_as_uint(f);
    unsigned int r = u + 0x7FFFu + ((u >> 16) & 1u);   // round-to-nearest-even
    return (u16)(r >> 16);
}

// ---------------------------------------------------------------------------
// CDNA5 async global -> LDS copy (ASYNCcnt-tracked), per-lane 16B.
// ---------------------------------------------------------------------------
__device__ __forceinline__ void async_load_b128(const void* gptr, void* lptr) {
    unsigned int ldsoff = (unsigned int)(size_t)lptr;   // low 32 bits = LDS byte offset
    asm volatile("global_load_async_to_lds_b128 %0, %1, off"
                 :: "v"(ldsoff), "v"(gptr) : "memory");
}
#define WAIT_ASYNCCNT(n) asm volatile("s_wait_asynccnt " #n ::: "memory")

// ---------------------------------------------------------------------------
// f32 -> bf16 elementwise conversion (grid-stride)
// ---------------------------------------------------------------------------
__global__ __launch_bounds__(256) void cvt_bf16(const float* __restrict__ in,
                                                u16* __restrict__ out, long n) {
    long i = (long)blockIdx.x * blockDim.x + threadIdx.x;
    long stride = (long)gridDim.x * blockDim.x;
    for (; i < n; i += stride) out[i] = f32_to_bf16(in[i]);
}

// ---------------------------------------------------------------------------
// C[M,N] = A[M,K] * B[N,K]^T   (A,B bf16 row-major, C f32)
// Block: 128 threads = 4 waves. Block tile 128(M) x 64(N); wave = 32(M) x 64(N)
// via 2x4 accumulators of v_wmma_f32_16x16x32_bf16.
// LDS tiles double-buffered and filled with global_load_async_to_lds_b128.
// ---------------------------------------------------------------------------
__global__ __launch_bounds__(128) void gemm_bf16_nt(const u16* __restrict__ A,
                                                    const u16* __restrict__ Bm,
                                                    float* __restrict__ C,
                                                    int M, int N, int K) {
    __shared__ u16 sA[2][128 * 40];   // 128 rows x 32 k, stride 40 (bank-conflict pad)
    __shared__ u16 sB[2][64 * 40];    // 64 n-rows x 32 k

    const int tid  = threadIdx.x;
    const int wave = tid >> 5;     // 0..3 -> 32-row M sub-tile
    const int lane = tid & 31;
    const int lh   = lane & 15;
    const int half = lane >> 4;    // 0: lanes 0-15, 1: lanes 16-31
    const int m0   = blockIdx.y * 128;
    const int n0   = blockIdx.x * 64;

    v8f acc[2][4];
#pragma unroll
    for (int mi = 0; mi < 2; ++mi)
#pragma unroll
        for (int j = 0; j < 4; ++j)
#pragma unroll
            for (int r = 0; r < 8; ++r) acc[mi][j][r] = 0.0f;

    // staging: A row per thread (64B = 4x b128), B half-row per thread (2x b128)
    const int arow = tid;                              // 0..127
    const int brow = tid >> 1, bks = (tid & 1) * 16;   // 0..63, {0,16}
    const u16* agp = &A[(size_t)(m0 + arow) * K];
    const u16* bgp = &Bm[(size_t)(n0 + brow) * K + bks];

    const int nk = K / 32;
    // stage slab 0 into buffer 0 (6 async b128 per thread)
#pragma unroll
    for (int c = 0; c < 4; ++c)
        async_load_b128(agp + c * 8, &sA[0][arow * 40 + c * 8]);
    async_load_b128(bgp,     &sB[0][brow * 40 + bks]);
    async_load_b128(bgp + 8, &sB[0][brow * 40 + bks + 8]);

    for (int ks = 0; ks < nk; ++ks) {
        const int bi = ks & 1;
        if (ks + 1 < nk) {
            const int k1 = (ks + 1) * 32, ni = (ks + 1) & 1;
#pragma unroll
            for (int c = 0; c < 4; ++c)
                async_load_b128(agp + k1 + c * 8, &sA[ni][arow * 40 + c * 8]);
            async_load_b128(bgp + k1,     &sB[ni][brow * 40 + bks]);
            async_load_b128(bgp + k1 + 8, &sB[ni][brow * 40 + bks + 8]);
            if (ks + 2 < nk) {  // warm L2 two slabs ahead
                __builtin_prefetch(agp + k1 + 32, 0, 3);
                __builtin_prefetch(bgp + k1 + 32, 0, 3);
            }
            WAIT_ASYNCCNT(6);   // previous slab resident, next still in flight
        } else {
            WAIT_ASYNCCNT(0);
        }
        __syncthreads();

        // ---- A fragments (16x32 bf16): lane = M row (lh), half selects K phase --
        // elems 0..7 -> K = half*8 .. +7 ; elems 8..15 -> K = half*8+16 .. +23
        union { v16bf v; uint4 q[2]; } ua[2];
#pragma unroll
        for (int mi = 0; mi < 2; ++mi) {
            const int m_loc = wave * 32 + mi * 16 + lh;
            ua[mi].q[0] = *(const uint4*)&sA[bi][m_loc * 40 + half * 8];
            ua[mi].q[1] = *(const uint4*)&sA[bi][m_loc * 40 + half * 8 + 16];
        }
#pragma unroll
        for (int j = 0; j < 4; ++j) {
            // ---- B fragment (32x16 bf16): lane = N col (lh), half selects K phase --
            union { v16bf v; uint4 q[2]; } ub;
            const int n_loc = j * 16 + lh;
            ub.q[0] = *(const uint4*)&sB[bi][n_loc * 40 + half * 8];
            ub.q[1] = *(const uint4*)&sB[bi][n_loc * 40 + half * 8 + 16];
#pragma unroll
            for (int mi = 0; mi < 2; ++mi)
                acc[mi][j] = __builtin_amdgcn_wmma_f32_16x16x32_bf16(
                    false, ua[mi].v, false, ub.v, (short)0, acc[mi][j], false, false);
        }
        __syncthreads();   // done reading buffer bi before it is refilled
    }

    // ---- store: VGPR r holds M = r + 8*half ; lane lh holds N ----
#pragma unroll
    for (int mi = 0; mi < 2; ++mi)
#pragma unroll
        for (int j = 0; j < 4; ++j) {
            const int n = n0 + j * 16 + lh;
#pragma unroll
            for (int r = 0; r < 8; ++r) {
                const int m = m0 + wave * 32 + mi * 16 + r + half * 8;
                if (m < M && n < N) C[(size_t)m * N + n] = acc[mi][j][r];
            }
        }
}

// ---------------------------------------------------------------------------
// Depthwise causal conv (K=4) + q/k l2norm + gating.
// Grid (NH, S, B), 64 threads (2 waves); thread = dim d within head.
// ---------------------------------------------------------------------------
__global__ __launch_bounds__(64) void conv_gate(const float* __restrict__ qkvz,
                                                const float* __restrict__ conv_w,
                                                const float* __restrict__ A_log,
                                                const float* __restrict__ dt_bias,
                                                float* __restrict__ qf,
                                                float* __restrict__ kf,
                                                float* __restrict__ vf,
                                                float* __restrict__ gg,
                                                float* __restrict__ bb) {
    const int h = blockIdx.x, s = blockIdx.y, b = blockIdx.z, d = threadIdx.x;

    const int cq = h * DKH + d;            // mixed channel ids
    const int ck = 2048 + cq;
    const int cv = 4096 + cq;
    const int colq = h * 256 + d;          // qkvz column ids
    const int colk = colq + 64;
    const int colv = colq + 128;

    float qr = 0.f, kr = 0.f, vr = 0.f;
#pragma unroll
    for (int j = 0; j < 4; ++j) {
        const int sp = s - 3 + j;
        if (sp >= 0) {
            const size_t row = (size_t)(b * SEQ + sp) * NQ;
            qr += conv_w[cq * 4 + j] * qkvz[row + colq];
            kr += conv_w[ck * 4 + j] * qkvz[row + colk];
            vr += conv_w[cv * 4 + j] * qkvz[row + colv];
        }
    }

    // l2-norm reductions across the 64 threads (intra-wave shuffle + LDS combine)
    float sq = qr * qr, sk = kr * kr;
#pragma unroll
    for (int off = 16; off > 0; off >>= 1) {
        sq += __shfl_xor(sq, off);
        sk += __shfl_xor(sk, off);
    }
    __shared__ float red[4];
    if ((threadIdx.x & 31) == 0) {
        red[(threadIdx.x >> 5) * 2]     = sq;
        red[(threadIdx.x >> 5) * 2 + 1] = sk;
    }
    __syncthreads();
    const float sumq = red[0] + red[2];
    const float sumk = red[1] + red[3];

    const float qv = qr * rsqrtf(sumq + EPSV) * 0.125f;   // * DK^-0.5
    const float kv = kr * rsqrtf(sumk + EPSV);

    const size_t t = ((size_t)(b * NHEAD + h) * SEQ + s) * DKH + d;  // time-major
    qf[t] = qv;
    kf[t] = kv;
    vf[t] = vr;

    if (threadIdx.x == 0) {
        const size_t row = (size_t)(b * SEQ + s) * NQ;
        const float bv = qkvz[row + 8192 + 2 * h];
        const float av = qkvz[row + 8192 + 2 * h + 1];
        const float x  = av + dt_bias[h];
        const float sp = fmaxf(x, 0.f) + log1pf(__expf(-fabsf(x)));  // softplus
        gg[(size_t)(b * NHEAD + h) * SEQ + s] = -__expf(A_log[h]) * sp;
        bb[(size_t)(b * NHEAD + h) * SEQ + s] = 1.f / (1.f + __expf(-bv));
    }
}

// ---------------------------------------------------------------------------
// Gated delta-rule scan. Grid (NH, B), 64 threads (2 waves).
// Lane owns one dv column; state column (64 dk values) lives in registers.
// q/k/v are staged in 32-step chunks via async global->LDS, double-buffered.
// ---------------------------------------------------------------------------
__global__ __launch_bounds__(64) void recurrence(const float* __restrict__ qf,
                                                 const float* __restrict__ kf,
                                                 const float* __restrict__ vf,
                                                 const float* __restrict__ gg,
                                                 const float* __restrict__ bb,
                                                 float* __restrict__ o) {
    const int h = blockIdx.x, b = blockIdx.y;
    const int dv = threadIdx.x;
    const size_t base = (size_t)(b * NHEAD + h) * SEQ * DKH;
    const float* qp = qf + base;
    const float* kp = kf + base;
    const float* vp = vf + base;
    const float* gp = gg + (size_t)(b * NHEAD + h) * SEQ;
    const float* bp = bb + (size_t)(b * NHEAD + h) * SEQ;

    float st[DKH];
#pragma unroll
    for (int i = 0; i < DKH; ++i) st[i] = 0.f;

    __shared__ float sbuf[2][3][CHUNK * DKH];   // q,k,v chunk (24KB per buffer)
    __shared__ float sg[2][CHUNK], sbt[2][CHUNK];

    // issue chunk 0 into buffer 0 (8 async b128 per tensor per thread)
#pragma unroll
    for (int i = 0; i < 8; ++i) {
        const int off = i * 256 + threadIdx.x * 4;
        async_load_b128(qp + off, &sbuf[0][0][off]);
        async_load_b128(kp + off, &sbuf[0][1][off]);
        async_load_b128(vp + off, &sbuf[0][2][off]);
    }
    if (threadIdx.x < CHUNK) {
        sg[0][threadIdx.x]  = gp[threadIdx.x];
        sbt[0][threadIdx.x] = bp[threadIdx.x];
    }

    const int nchunk = SEQ / CHUNK;
    for (int c = 0; c < nchunk; ++c) {
        const int bi = c & 1;
        if (c + 1 < nchunk) {
            const int ni = (c + 1) & 1;
            const size_t nf = (size_t)(c + 1) * CHUNK * DKH;
#pragma unroll
            for (int i = 0; i < 8; ++i) {
                const int off = i * 256 + threadIdx.x * 4;
                async_load_b128(qp + nf + off, &sbuf[ni][0][off]);
                async_load_b128(kp + nf + off, &sbuf[ni][1][off]);
                async_load_b128(vp + nf + off, &sbuf[ni][2][off]);
            }
            if (threadIdx.x < CHUNK) {
                sg[ni][threadIdx.x]  = gp[(c + 1) * CHUNK + threadIdx.x];
                sbt[ni][threadIdx.x] = bp[(c + 1) * CHUNK + threadIdx.x];
            }
            WAIT_ASYNCCNT(24);   // chunk c resident, chunk c+1 in flight
        } else {
            WAIT_ASYNCCNT(0);
        }
        __syncthreads();

        const float* bq = sbuf[bi][0];
        const float* bk = sbuf[bi][1];
        const float* bv = sbuf[bi][2];
        for (int s = 0; s < CHUNK; ++s) {
            const float dec  = __expf(sg[bi][s]);
            const float beta = sbt[bi][s];

            float kvdot = 0.f;
#pragma unroll
            for (int i = 0; i < DKH; ++i) {
                st[i] *= dec;
                kvdot = fmaf(bk[s * DKH + i], st[i], kvdot);
            }
            const float delta = (bv[s * DKH + dv] - kvdot) * beta;

            float oacc = 0.f;
#pragma unroll
            for (int i = 0; i < DKH; ++i) {
                st[i] = fmaf(bk[s * DKH + i], delta, st[i]);
                oacc  = fmaf(bq[s * DKH + i], st[i], oacc);
            }
            o[((size_t)(b * SEQ + c * CHUNK + s) * NHEAD + h) * DVH + dv] = oacc;
        }
        __syncthreads();   // all waves done with buffer bi before refill
    }
}

// ---------------------------------------------------------------------------
// Gated RMSNorm -> bf16 A-matrix for the output GEMM. Grid (NH, S, B), 64 thr.
// ---------------------------------------------------------------------------
__global__ __launch_bounds__(64) void rmsnorm_gate(const float* __restrict__ o,
                                                   const float* __restrict__ qkvz,
                                                   const float* __restrict__ norm_w,
                                                   u16* __restrict__ a2) {
    const int h = blockIdx.x, s = blockIdx.y, b = blockIdx.z, d = threadIdx.x;
    const float ov = o[((size_t)(b * SEQ + s) * NHEAD + h) * DVH + d];

    float ss = ov * ov;
#pragma unroll
    for (int off = 16; off > 0; off >>= 1) ss += __shfl_xor(ss, off);
    __shared__ float red[2];
    if ((threadIdx.x & 31) == 0) red[threadIdx.x >> 5] = ss;
    __syncthreads();
    const float var = (red[0] + red[1]) * (1.0f / DVH);

    const float z   = qkvz[(size_t)(b * SEQ + s) * NQ + h * 256 + 192 + d];
    const float sig = 1.f / (1.f + __expf(-z));
    const float res = ov * rsqrtf(var + EPSV) * norm_w[d] * sig;
    a2[(size_t)(b * SEQ + s) * HIDN + h * DVH + d] = f32_to_bf16(res);
}

// ---------------------------------------------------------------------------
extern "C" void kernel_launch(void* const* d_in, const int* in_sizes, int n_in,
                              void* d_out, int out_size, void* d_ws, size_t ws_size,
                              hipStream_t stream) {
    const float* hidden  = (const float*)d_in[0];
    const float* w_qkvz  = (const float*)d_in[1];
    const float* w_ba    = (const float*)d_in[2];
    const float* conv_w  = (const float*)d_in[3];
    const float* A_log   = (const float*)d_in[4];
    const float* dt_bias = (const float*)d_in[5];
    const float* norm_w  = (const float*)d_in[6];
    const float* w_out   = (const float*)d_in[7];
    float* out = (float*)d_out;

    char* ws = (char*)d_ws;
    size_t off = 0;
    auto carve = [&](size_t bytes) -> char* {
        char* p = ws + off;
        off += (bytes + 255) & ~(size_t)255;
        return p;
    };
    u16*   hbf   = (u16*)  carve((size_t)MROWS * HIDN * 2);        // hidden bf16
    u16*   wcbf  = (u16*)  carve((size_t)NQ * HIDN * 2);           // [w_qkvz ; w_ba] bf16
    u16*   wobf  = (u16*)  carve((size_t)HIDN * HIDN * 2);         // w_out bf16
    float* qkvzf = (float*)carve((size_t)MROWS * NQ * 4);          // projection output
    float* qf    = (float*)carve((size_t)MROWS * HIDN * 4);
    float* kf    = (float*)carve((size_t)MROWS * HIDN * 4);
    float* vf    = (float*)carve((size_t)MROWS * HIDN * 4);
    float* gg    = (float*)carve((size_t)MROWS * NHEAD * 4);
    float* bb    = (float*)carve((size_t)MROWS * NHEAD * 4);
    float* orw   = (float*)carve((size_t)MROWS * HIDN * 4);
    u16*   a2    = (u16*)  carve((size_t)MROWS * HIDN * 2);
    (void)ws_size; (void)in_sizes; (void)n_in; (void)out_size;

    // 1) bf16 conversions
    cvt_bf16<<<4096, 256, 0, stream>>>(hidden, hbf, (long)MROWS * HIDN);
    cvt_bf16<<<4096, 256, 0, stream>>>(w_qkvz, wcbf, (long)8192 * HIDN);
    cvt_bf16<<<512,  256, 0, stream>>>(w_ba, wcbf + (size_t)8192 * HIDN, (long)64 * HIDN);
    cvt_bf16<<<4096, 256, 0, stream>>>(w_out, wobf, (long)HIDN * HIDN);

    // 2) fused qkvz + ba projection: [8192,2048] x [8256,2048]^T
    gemm_bf16_nt<<<dim3(NQ / 64, MROWS / 128), 128, 0, stream>>>(
        hbf, wcbf, qkvzf, MROWS, NQ, HIDN);

    // 3) conv + norms + gating
    conv_gate<<<dim3(NHEAD, SEQ, BSZ), 64, 0, stream>>>(
        qkvzf, conv_w, A_log, dt_bias, qf, kf, vf, gg, bb);

    // 4) sequential gated delta-rule scan (async-chunked)
    recurrence<<<dim3(NHEAD, BSZ), 64, 0, stream>>>(qf, kf, vf, gg, bb, orw);

    // 5) gated RMSNorm -> bf16
    rmsnorm_gate<<<dim3(NHEAD, SEQ, BSZ), 64, 0, stream>>>(orw, qkvzf, norm_w, a2);

    // 6) output projection: [8192,2048] x [2048,2048]^T -> d_out (f32)
    gemm_bf16_nt<<<dim3(HIDN / 64, MROWS / 128), 128, 0, stream>>>(
        a2, wobf, out, MROWS, HIDN, HIDN);
}
